// HyperDAGEncoder_36670430773459
// MI455X (gfx1250) — matrix operand order repaired and checked by reading, not compiled, not run on hardware
//
#include <hip/hip_runtime.h>
#include <hip/hip_bf16.h>

// ---------------------------------------------------------------------------
// HyperDAG encoder for MI455X (gfx1250, wave32, WMMA 16x16x32 f16).
// GEMMs and attention run on v_wmma_f32_16x16x32_f16 with double-buffered
// LDS software pipelining: next tile's global_load_b128s issue before the
// current tile's WMMAs, converts/stores go to the alternate buffer, one
// barrier per K-step.  Attention is flash-style (online softmax), one
// barrier per 32-key chunk (P slab is wave-private; per-wave LDS ordering
// via DScnt).
// ---------------------------------------------------------------------------

#define B_  4
#define N_  1024
#define D_  512
#define H_  8
#define DH_ 64
#define L_  3
#define E_  2048
#define A_  8

typedef __attribute__((ext_vector_type(16))) _Float16 v16h;
typedef __attribute__((ext_vector_type(8)))  float    v8f;

union V16H {
    v16h v;
    _Float16 h[16];
    uint4 q[2];
};
union H4 { _Float16 h[4]; uint2 u; };
union H2 { _Float16 h[2]; unsigned u; };

__device__ __forceinline__ float gelu_f(float x) {
    return 0.5f * x * (1.0f + erff(x * 0.70710678118654752f));
}

// ---------------------------------------------------------------------------
// Generic WMMA GEMM: C[M,N] = act(A[M,K] @ W[K,N] + bias[N])
// Block tile 128x64, BK=32, 256 threads = 8 waves in a 4x2 grid of 32x32
// wave tiles.  A staged row-major f16, W staged transposed ([n][k]) so both
// fragment loads are two aligned ds_load_b128 matching the 16-bit A/B-matrix
// VGPR layouts (lanes 0-15: K {0..7,16..23}; lanes 16-31: K {8..15,24..31}).
// Double-buffered LDS; loads for tile kt+1 issue before WMMAs of tile kt.
// ---------------------------------------------------------------------------
#define BM 128
#define BN 64
#define BK 32
#define LSTR 40   // halves; multiple of 8 -> 16B-aligned fragment segments

__launch_bounds__(256)
__global__ void wmma_gemm_kernel(const float* __restrict__ A,
                                 const float* __restrict__ W,
                                 const float* __restrict__ bias,
                                 float* __restrict__ C,
                                 int M, int K, int N, int act) {
    __shared__ __align__(16) _Float16 As[2 * BM * LSTR];
    __shared__ __align__(16) _Float16 Bs[2 * BN * LSTR];

    const int tid  = threadIdx.x;
    const int lane = tid & 31;
    const int l16  = lane & 15;
    const int hi   = lane >> 4;          // 0 or 1
    const int w    = tid >> 5;           // wave id 0..7
    const int wm   = w & 3;              // 4 m-slots of 32 rows
    const int wn   = w >> 2;             // 2 n-slots of 32 cols
    const int bMr  = (int)blockIdx.x * BM;
    const int bNc  = (int)blockIdx.y * BN;

    // Fixed staging coordinates (no per-element predication anywhere).
    const int sa_m  = tid >> 3;          // A: rows 0..31 (+i*32), 8 float4/row
    const int sa_k4 = tid & 7;
    const int sb_k  = (tid >> 4) * 2;    // B: k pair
    const int sb_n  = (tid & 15) * 4;    // B: 4 consecutive n

    float4 ra[4], rbA, rbB;              // staging register buffers

    auto load_tile = [&](int k0) {
#pragma unroll
        for (int i = 0; i < 4; ++i) {
            int row = bMr + i * 32 + sa_m;
            if (row >= M) row = M - 1;   // clamp: OOB rows never stored
            ra[i] = *reinterpret_cast<const float4*>(
                &A[(size_t)row * K + k0 + sa_k4 * 4]);
        }
        const float* w0 = &W[(size_t)(k0 + sb_k) * N + bNc + sb_n];
        rbA = *reinterpret_cast<const float4*>(w0);
        rbB = *reinterpret_cast<const float4*>(w0 + N);
    };
    auto store_tile = [&](int sel) {
        _Float16* Asb = &As[sel * (BM * LSTR)];
        _Float16* Bsb = &Bs[sel * (BN * LSTR)];
#pragma unroll
        for (int i = 0; i < 4; ++i) {
            H4 pk;
            pk.h[0] = (_Float16)ra[i].x; pk.h[1] = (_Float16)ra[i].y;
            pk.h[2] = (_Float16)ra[i].z; pk.h[3] = (_Float16)ra[i].w;
            *reinterpret_cast<uint2*>(&Asb[(i * 32 + sa_m) * LSTR + sa_k4 * 4]) = pk.u;
        }
        H2 p0, p1, p2, p3;
        p0.h[0] = (_Float16)rbA.x; p0.h[1] = (_Float16)rbB.x;
        p1.h[0] = (_Float16)rbA.y; p1.h[1] = (_Float16)rbB.y;
        p2.h[0] = (_Float16)rbA.z; p2.h[1] = (_Float16)rbB.z;
        p3.h[0] = (_Float16)rbA.w; p3.h[1] = (_Float16)rbB.w;
        *reinterpret_cast<unsigned*>(&Bsb[(sb_n + 0) * LSTR + sb_k]) = p0.u;
        *reinterpret_cast<unsigned*>(&Bsb[(sb_n + 1) * LSTR + sb_k]) = p1.u;
        *reinterpret_cast<unsigned*>(&Bsb[(sb_n + 2) * LSTR + sb_k]) = p2.u;
        *reinterpret_cast<unsigned*>(&Bsb[(sb_n + 3) * LSTR + sb_k]) = p3.u;
    };

    v8f acc[2][2];
#pragma unroll
    for (int i = 0; i < 2; ++i)
#pragma unroll
        for (int j = 0; j < 2; ++j)
            acc[i][j] = (v8f)(0.0f);

    const int nK = K / BK;
    load_tile(0);
    store_tile(0);
    __syncthreads();

    for (int kt = 0; kt < nK; ++kt) {
        const int cur = kt & 1;
        if (kt + 1 < nK) {
            load_tile((kt + 1) * BK);    // overlap with WMMAs below
            if (kt + 2 < nK) {
                int pr = bMr + (tid >> 1);
                if (pr >= M) pr = M - 1;
                __builtin_prefetch(&A[(size_t)pr * K + (kt + 2) * BK + (tid & 1) * 16], 0, 1);
                __builtin_prefetch(&W[(size_t)((kt + 2) * BK + (tid >> 6)) * N + bNc + (tid & 63)], 0, 1);
            }
        }
        const _Float16* Asb = &As[cur * (BM * LSTR)];
        const _Float16* Bsb = &Bs[cur * (BN * LSTR)];

        V16H af[2], bf[2];
#pragma unroll
        for (int ms = 0; ms < 2; ++ms) {
            const _Float16* p = &Asb[(wm * 32 + ms * 16 + l16) * LSTR + hi * 8];
            af[ms].q[0] = *reinterpret_cast<const uint4*>(p);       // K base+0..7
            af[ms].q[1] = *reinterpret_cast<const uint4*>(p + 16);  // K base+16..23
        }
#pragma unroll
        for (int ns = 0; ns < 2; ++ns) {
            const _Float16* p = &Bsb[(wn * 32 + ns * 16 + l16) * LSTR + hi * 8];
            bf[ns].q[0] = *reinterpret_cast<const uint4*>(p);       // K base+0..7
            bf[ns].q[1] = *reinterpret_cast<const uint4*>(p + 16);  // K base+16..23
        }
#pragma unroll
        for (int ms = 0; ms < 2; ++ms)
#pragma unroll
            for (int ns = 0; ns < 2; ++ns)
                acc[ms][ns] = __builtin_amdgcn_wmma_f32_16x16x32_f16(
                    false, af[ms].v, false, bf[ns].v,
                    (short)0, acc[ms][ns], false, false);

        if (kt + 1 < nK) store_tile(1 - cur);   // other buffer: no conflict
        __syncthreads();
    }

    // ---- epilogue: bias + activation, C-matrix layout stores ----
#pragma unroll
    for (int ms = 0; ms < 2; ++ms) {
#pragma unroll
        for (int ns = 0; ns < 2; ++ns) {
            int row0 = bMr + wm * 32 + ms * 16 + hi * 8;
            int col  = bNc + wn * 32 + ns * 16 + l16;
            float bv = bias ? bias[col] : 0.0f;
#pragma unroll
            for (int r = 0; r < 8; ++r) {
                int row = row0 + r;
                if (row < M) {
                    float v = acc[ms][ns][r] + bv;
                    if (act == 1) v = gelu_f(v);
                    C[(size_t)row * N + col] = v;
                }
            }
        }
    }
}

// ---------------------------------------------------------------------------
// Flash attention: 4 waves per block, each wave owns 16 query rows of a
// 64-row tile for one (b,h).  Keys streamed in 32-wide chunks with
// double-buffered LDS (K as [key][dh], V transposed [dh][key]); next chunk's
// global loads issue before the S-phase WMMAs.  Online softmax with 16-lane
// shfl_xor row reductions; P goes through a wave-private LDS slab (no
// barrier needed: per-wave LDS ordering).  One barrier per chunk.
// ---------------------------------------------------------------------------
#define VSTR 40

__launch_bounds__(128)
__global__ void flash_attn_kernel(const float* __restrict__ q,
                                  const float* __restrict__ k,
                                  const float* __restrict__ v,
                                  const unsigned char* __restrict__ nmask,
                                  float* __restrict__ o) {
    __shared__ __align__(16) _Float16 Ks[2 * 32 * VSTR];   // [key][dh]
    __shared__ __align__(16) _Float16 Vs[2 * 64 * VSTR];   // [dh][key]
    __shared__ __align__(16) _Float16 Ps[4 * 16 * 32];     // per-wave P

    const int tid  = threadIdx.x;
    const int lane = tid & 31;
    const int wv   = tid >> 5;     // wave 0..3
    const int l16  = lane & 15;
    const int hi   = lane >> 4;
    const int qtb  = blockIdx.x;   // 64-query-row tile
    const int h    = blockIdx.y;
    const int b    = blockIdx.z;
    const size_t base = (size_t)b * N_ * D_;
    const int hoff = h * DH_;

    // Q A-fragments for this wave's 16 rows (two K=32 fragments)
    V16H qa[2];
    {
        const float* qp = q + base + (size_t)(qtb * 64 + wv * 16 + l16) * D_ + hoff;
#pragma unroll
        for (int kf = 0; kf < 2; ++kf) {
            int d0 = kf * 32 + hi * 8;
            float4 f0 = *reinterpret_cast<const float4*>(qp + d0);
            float4 f1 = *reinterpret_cast<const float4*>(qp + d0 + 4);
            float4 f2 = *reinterpret_cast<const float4*>(qp + d0 + 16);
            float4 f3 = *reinterpret_cast<const float4*>(qp + d0 + 20);
            qa[kf].h[0]  = (_Float16)f0.x; qa[kf].h[1]  = (_Float16)f0.y;
            qa[kf].h[2]  = (_Float16)f0.z; qa[kf].h[3]  = (_Float16)f0.w;
            qa[kf].h[4]  = (_Float16)f1.x; qa[kf].h[5]  = (_Float16)f1.y;
            qa[kf].h[6]  = (_Float16)f1.z; qa[kf].h[7]  = (_Float16)f1.w;
            qa[kf].h[8]  = (_Float16)f2.x; qa[kf].h[9]  = (_Float16)f2.y;
            qa[kf].h[10] = (_Float16)f2.z; qa[kf].h[11] = (_Float16)f2.w;
            qa[kf].h[12] = (_Float16)f3.x; qa[kf].h[13] = (_Float16)f3.y;
            qa[kf].h[14] = (_Float16)f3.z; qa[kf].h[15] = (_Float16)f3.w;
        }
    }

    float m_i[8], l_i[8];
    v8f oacc[4];
#pragma unroll
    for (int r = 0; r < 8; ++r) { m_i[r] = -INFINITY; l_i[r] = 0.0f; }
#pragma unroll
    for (int d = 0; d < 4; ++d) oacc[d] = (v8f)(0.0f);
    _Float16* myPs = &Ps[wv * 512];

    const int skey = tid & 31;        // staging: key row
    const int sd0  = (tid >> 5) * 16; // staging: 16-dh slice

    float4 fk[4], fv[4];
    auto load_chunk = [&](int ch) {
        const float* kp = k + base + (size_t)(ch * 32 + skey) * D_ + hoff + sd0;
        const float* vp = v + base + (size_t)(ch * 32 + skey) * D_ + hoff + sd0;
#pragma unroll
        for (int j = 0; j < 4; ++j) {
            fk[j] = *reinterpret_cast<const float4*>(kp + j * 4);
            fv[j] = *reinterpret_cast<const float4*>(vp + j * 4);
        }
    };
    auto store_chunk = [&](int sel) {
        _Float16* Ksb = &Ks[sel * (32 * VSTR)];
        _Float16* Vsb = &Vs[sel * (64 * VSTR)];
#pragma unroll
        for (int j = 0; j < 4; ++j) {
            H4 pk;
            pk.h[0] = (_Float16)fk[j].x; pk.h[1] = (_Float16)fk[j].y;
            pk.h[2] = (_Float16)fk[j].z; pk.h[3] = (_Float16)fk[j].w;
            *reinterpret_cast<uint2*>(&Ksb[skey * VSTR + sd0 + j * 4]) = pk.u;
            Vsb[(sd0 + j * 4 + 0) * VSTR + skey] = (_Float16)fv[j].x;
            Vsb[(sd0 + j * 4 + 1) * VSTR + skey] = (_Float16)fv[j].y;
            Vsb[(sd0 + j * 4 + 2) * VSTR + skey] = (_Float16)fv[j].z;
            Vsb[(sd0 + j * 4 + 3) * VSTR + skey] = (_Float16)fv[j].w;
        }
    };

    const int nCh = N_ / 32;
    load_chunk(0);
    store_chunk(0);

    for (int ch = 0; ch < nCh; ++ch) {
        const int cur = ch & 1;
        __syncthreads();   // staged chunk `cur` visible to all waves
        if (ch + 1 < nCh) load_chunk(ch + 1);   // overlap with S phase

        const _Float16* Ksb = &Ks[cur * (32 * VSTR)];
        const _Float16* Vsb = &Vs[cur * (64 * VSTR)];

        // ---- S chunk (16x32) = Q(16x64) @ K^T(64x32) ----
        v8f s[2];
        float maskc[2];
#pragma unroll
        for (int cf = 0; cf < 2; ++cf) {
            s[cf] = (v8f)(0.0f);
            maskc[cf] = nmask[b * N_ + ch * 32 + cf * 16 + l16] ? 1.0f : 0.0f;
#pragma unroll
            for (int kf = 0; kf < 2; ++kf) {
                V16H kb;
                const _Float16* p = &Ksb[(cf * 16 + l16) * VSTR + kf * 32 + hi * 8];
                kb.q[0] = *reinterpret_cast<const uint4*>(p);
                kb.q[1] = *reinterpret_cast<const uint4*>(p + 16);
                s[cf] = __builtin_amdgcn_wmma_f32_16x16x32_f16(
                    false, qa[kf].v, false, kb.v, (short)0, s[cf], false, false);
            }
        }
        // ---- scale, mask, online softmax ----
        float cm[8];
#pragma unroll
        for (int r = 0; r < 8; ++r) {
            float t0 = maskc[0] ? s[0][r] * 0.125f : -INFINITY;
            float t1 = maskc[1] ? s[1][r] * 0.125f : -INFINITY;
            s[0][r] = t0; s[1][r] = t1;
            cm[r] = fmaxf(t0, t1);
        }
#pragma unroll
        for (int off = 1; off < 16; off <<= 1)
#pragma unroll
            for (int r = 0; r < 8; ++r)
                cm[r] = fmaxf(cm[r], __shfl_xor(cm[r], off));

        float mnew[8], scl[8], cs[8];
#pragma unroll
        for (int r = 0; r < 8; ++r) {
            mnew[r] = fmaxf(m_i[r], cm[r]);
            scl[r]  = (m_i[r] == -INFINITY) ? 0.0f : expf(m_i[r] - mnew[r]);
            float p0 = (s[0][r] == -INFINITY) ? 0.0f : expf(s[0][r] - mnew[r]);
            float p1 = (s[1][r] == -INFINITY) ? 0.0f : expf(s[1][r] - mnew[r]);
            cs[r] = p0 + p1;
            int row = r + hi * 8;
            myPs[row * 32 + l16]      = (_Float16)p0;
            myPs[row * 32 + 16 + l16] = (_Float16)p1;
        }
#pragma unroll
        for (int off = 1; off < 16; off <<= 1)
#pragma unroll
            for (int r = 0; r < 8; ++r)
                cs[r] += __shfl_xor(cs[r], off);
#pragma unroll
        for (int r = 0; r < 8; ++r) {
            l_i[r] = l_i[r] * scl[r] + cs[r];
            m_i[r] = mnew[r];
        }
#pragma unroll
        for (int d = 0; d < 4; ++d)
#pragma unroll
            for (int r = 0; r < 8; ++r)
                oacc[d][r] *= scl[r];

        if (ch + 1 < nCh) store_chunk(1 - cur);   // other buffer: no conflict

        // ---- O += P(16x32) @ V_chunk(32x64) ----
        // myPs is wave-private; per-wave LDS ops complete in order (DScnt),
        // so no barrier is required between the stores above and these loads.
        V16H pa;
        {
            const _Float16* pp = &myPs[l16 * 32 + hi * 8];
            pa.q[0] = *reinterpret_cast<const uint4*>(pp);       // K base+0..7
            pa.q[1] = *reinterpret_cast<const uint4*>(pp + 16);  // K base+16..23
        }
#pragma unroll
        for (int d = 0; d < 4; ++d) {
            V16H vbf;
            const _Float16* p = &Vsb[(d * 16 + l16) * VSTR + hi * 8];
            vbf.q[0] = *reinterpret_cast<const uint4*>(p);
            vbf.q[1] = *reinterpret_cast<const uint4*>(p + 16);
            oacc[d] = __builtin_amdgcn_wmma_f32_16x16x32_f16(
                false, pa.v, false, vbf.v, (short)0, oacc[d], false, false);
        }
    }

    // ---- write O / l ----
#pragma unroll
    for (int d = 0; d < 4; ++d) {
#pragma unroll
        for (int r = 0; r < 8; ++r) {
            float denom = (l_i[r] > 0.0f) ? l_i[r] : 1.0f;
            int row = qtb * 64 + wv * 16 + hi * 8 + r;
            o[base + (size_t)row * D_ + hoff + d * 16 + l16] = oacc[d][r] / denom;
        }
    }
}

// ---------------------------------------------------------------------------
// LayerNorm over 512 features, optional residual add and pre-GELU.
// ---------------------------------------------------------------------------
__launch_bounds__(256)
__global__ void ln512_kernel(float* __restrict__ out, const float* __restrict__ in,
                             const float* __restrict__ res,
                             const float* __restrict__ g, const float* __restrict__ bt,
                             int gelu_first) {
    __shared__ float sm[256], sv[256];
    const int row = blockIdx.x, t = threadIdx.x;
    const size_t base = (size_t)row * 512;
    float x0 = in[base + t], x1 = in[base + 256 + t];
    if (res) { x0 += res[base + t]; x1 += res[base + 256 + t]; }
    if (gelu_first) { x0 = gelu_f(x0); x1 = gelu_f(x1); }
    sm[t] = x0 + x1;
    sv[t] = x0 * x0 + x1 * x1;
    __syncthreads();
    for (int s = 128; s > 0; s >>= 1) {
        if (t < s) { sm[t] += sm[t + s]; sv[t] += sv[t + s]; }
        __syncthreads();
    }
    float mean = sm[0] * (1.0f / 512.0f);
    float var  = sv[0] * (1.0f / 512.0f) - mean * mean;
    float inv  = rsqrtf(var + 1e-5f);
    out[base + t]       = (x0 - mean) * inv * g[t] + bt[t];
    out[base + 256 + t] = (x1 - mean) * inv * g[256 + t] + bt[256 + t];
}

// ---------------------------------------------------------------------------
__global__ void embed_kernel(const int* __restrict__ ids,
                             const float* __restrict__ table,
                             float* __restrict__ x) {
    const int row = blockIdx.x, t = threadIdx.x;
    int id = ids[row] & 31;
    x[(size_t)row * 512 + t]       = table[id * 512 + t];
    x[(size_t)row * 512 + 256 + t] = table[id * 512 + 256 + t];
}

// pooled hyperedge features + edge-type embedding concat -> (B*E, 544)
__launch_bounds__(64)
__global__ void hyperedge_pool_kernel(const float* __restrict__ x,
                                      const int* __restrict__ members,
                                      const int* __restrict__ etypes,
                                      const unsigned char* __restrict__ mask,
                                      const float* __restrict__ et_emb,
                                      float* __restrict__ ef_in) {
    const int be = blockIdx.x;
    const int b = be / E_, e = be % E_;
    const int t = threadIdx.x;
    __shared__ int   mem_s[A_];
    __shared__ float mk[A_];
    if (t < A_) {
        int m = members[((size_t)b * E_ + e) * A_ + t];
        m = m < 0 ? 0 : (m > N_ - 1 ? N_ - 1 : m);
        mem_s[t] = m;
        mk[t] = mask[((size_t)b * E_ + e) * A_ + t] ? 1.0f : 0.0f;
    }
    __syncthreads();
    float c = 0.0f;
#pragma unroll
    for (int a = 0; a < A_; ++a) c += mk[a];
    c = fmaxf(c, 1.0f);
    const size_t orow = (size_t)be * 544;
    for (int col = t; col < D_; col += 64) {
        float acc = 0.0f;
#pragma unroll
        for (int a = 0; a < A_; ++a)
            acc += mk[a] * x[((size_t)b * N_ + mem_s[a]) * D_ + col];
        ef_in[orow + col] = acc / c;
    }
    if (t < 32) {
        int et = etypes[(size_t)b * E_ + e] & 15;
        ef_in[orow + 512 + t] = et_emb[et * 32 + t];
    }
}

// scatter edge features back to nodes (atomic segment sum)
__launch_bounds__(128)
__global__ void scatter_kernel(const float* __restrict__ ef,
                               const int* __restrict__ members,
                               const unsigned char* __restrict__ mask,
                               float* __restrict__ node_upd,
                               float* __restrict__ counts) {
    const int idx = blockIdx.x;             // b*E*A + e*A + a
    const int a = idx % A_;
    const int e = (idx / A_) % E_;
    const int b = idx / (A_ * E_);
    if (!mask[((size_t)b * E_ + e) * A_ + a]) return;
    int m = members[((size_t)b * E_ + e) * A_ + a];
    m = m < 0 ? 0 : (m > N_ - 1 ? N_ - 1 : m);
    const size_t src = ((size_t)b * E_ + e) * D_;
    const size_t dst = ((size_t)b * N_ + m) * D_;
    for (int col = threadIdx.x; col < D_; col += 128)
        atomicAdd(&node_upd[dst + col], ef[src + col]);
    if (threadIdx.x == 0)
        atomicAdd(&counts[(size_t)b * N_ + m], 1.0f);
}

// upd_in = concat(x, node_upd / clip(counts,1))  -> (B*N, 1024)
__launch_bounds__(256)
__global__ void build_upd_in_kernel(const float* __restrict__ x,
                                    const float* __restrict__ node_upd,
                                    const float* __restrict__ counts,
                                    float* __restrict__ upd_in) {
    const int row = blockIdx.x;
    const float c = fmaxf(counts[row], 1.0f);
    const size_t xo = (size_t)row * D_;
    const size_t uo = (size_t)row * (2 * D_);
    for (int col = threadIdx.x; col < D_; col += 256) {
        upd_in[uo + col]       = x[xo + col];
        upd_in[uo + 512 + col] = node_upd[xo + col] / c;
    }
}

// masked mean over nodes -> (B, D)
__launch_bounds__(256)
__global__ void graph_pool_kernel(const float* __restrict__ x,
                                  const unsigned char* __restrict__ nmask,
                                  float* __restrict__ gp) {
    const int b = blockIdx.x, t = threadIdx.x;
    float a0 = 0.0f, a1 = 0.0f, c = 0.0f;
    for (int n = 0; n < N_; ++n) {
        float m = nmask[b * N_ + n] ? 1.0f : 0.0f;
        c += m;
        a0 += m * x[((size_t)b * N_ + n) * D_ + t];
        a1 += m * x[((size_t)b * N_ + n) * D_ + 256 + t];
    }
    c = fmaxf(c, 1.0f);
    gp[(size_t)b * D_ + t]       = a0 / c;
    gp[(size_t)b * D_ + 256 + t] = a1 / c;
}

__global__ void zero_kernel(float* __restrict__ p, size_t n) {
    for (size_t i = (size_t)blockIdx.x * blockDim.x + threadIdx.x; i < n;
         i += (size_t)gridDim.x * blockDim.x)
        p[i] = 0.0f;
}

__global__ void copy_kernel(float* __restrict__ dst, const float* __restrict__ src, size_t n) {
    for (size_t i = (size_t)blockIdx.x * blockDim.x + threadIdx.x; i < n;
         i += (size_t)gridDim.x * blockDim.x)
        dst[i] = src[i];
}

// ---------------------------------------------------------------------------
extern "C" void kernel_launch(void* const* d_in, const int* in_sizes, int n_in,
                              void* d_out, int out_size, void* d_ws, size_t ws_size,
                              hipStream_t stream) {
    (void)in_sizes; (void)n_in; (void)out_size; (void)ws_size;

    // Inputs in setup_inputs() insertion order (params flattened depth-first).
    const int* node_type_ids = (const int*)d_in[0];
    const int* he_members    = (const int*)d_in[3];
    const int* he_types      = (const int*)d_in[4];
    const unsigned char* he_mask = (const unsigned char*)d_in[5];
    const unsigned char* nmask   = (const unsigned char*)d_in[6];
    const float* nte  = (const float*)d_in[7];
    const float* np_w = (const float*)d_in[8];  const float* np_b = (const float*)d_in[9];
    const float* q_w  = (const float*)d_in[10]; const float* q_b  = (const float*)d_in[11];
    const float* k_w  = (const float*)d_in[12]; const float* k_b  = (const float*)d_in[13];
    const float* v_w  = (const float*)d_in[14]; const float* v_b  = (const float*)d_in[15];
    const float* o_w  = (const float*)d_in[16]; const float* o_b  = (const float*)d_in[17];
    const float* n1_g = (const float*)d_in[18]; const float* n1_b = (const float*)d_in[19];
    const float* f1_w = (const float*)d_in[20]; const float* f1_b = (const float*)d_in[21];
    const float* f2_w = (const float*)d_in[22]; const float* f2_b = (const float*)d_in[23];
    const float* n2_g = (const float*)d_in[24]; const float* n2_b = (const float*)d_in[25];
    const float* et_emb = (const float*)d_in[26];
    const float* enc_w = (const float*)d_in[27]; const float* enc_b  = (const float*)d_in[28];
    const float* enc_g = (const float*)d_in[29]; const float* enc_bt = (const float*)d_in[30];
    const float* upd_w = (const float*)d_in[31]; const float* upd_b  = (const float*)d_in[32];
    const float* upd_g = (const float*)d_in[33]; const float* upd_bt = (const float*)d_in[34];
    const float* pool_w = (const float*)d_in[35]; const float* pool_b  = (const float*)d_in[36];
    const float* pool_g = (const float*)d_in[37]; const float* pool_bt = (const float*)d_in[38];

    float* ws = (float*)d_ws;
    size_t off = 0;
    auto alloc = [&](size_t n) { float* p = ws + off; off += n; return p; };
    const size_t BND = (size_t)B_ * N_ * D_;

    float* x   = alloc(BND);                       // activations
    float* h   = alloc(BND);                       // np output / x1
    float* qb  = alloc(BND);
    float* kb  = alloc(BND);
    float* vb  = alloc(BND);
    float* ao  = alloc(BND);                       // attention output
    float* t0  = alloc(BND);                       // o-proj / ff output
    float* ffh = alloc((size_t)B_ * N_ * 4 * D_);  // FFN hidden
    float* efi = alloc((size_t)B_ * E_ * 544);     // pooled ++ edge emb
    float* efl = alloc((size_t)B_ * E_ * D_);      // enc linear
    float* ef  = alloc((size_t)B_ * E_ * D_);      // edge features
    float* nup = alloc(BND);                       // node updates
    float* cnt = alloc((size_t)B_ * N_);
    float* ui  = alloc((size_t)B_ * N_ * 2 * D_);  // concat(x, node_upd)
    float* ul  = alloc(BND);                       // upd linear
    float* gp  = alloc((size_t)B_ * D_);
    float* gp2 = alloc((size_t)B_ * D_);

    auto gemm = [&](const float* Ap, const float* Wp, const float* bp, float* Cp,
                    int M, int K, int N, int act) {
        dim3 g((unsigned)((M + BM - 1) / BM), (unsigned)(N / BN));
        wmma_gemm_kernel<<<g, 256, 0, stream>>>(Ap, Wp, bp, Cp, M, K, N, act);
    };
    auto ln = [&](float* outp, const float* inp, const float* resp,
                  const float* gp_, const float* bp_, int rows, int gfirst) {
        ln512_kernel<<<dim3((unsigned)rows), 256, 0, stream>>>(outp, inp, resp, gp_, bp_, gfirst);
    };

    // x = node_type_embed[node_type_ids]
    embed_kernel<<<dim3(B_ * N_), 256, 0, stream>>>(node_type_ids, nte, x);

    const int M = B_ * N_;
    for (int l = 0; l < L_; ++l) {
        const size_t wDD = (size_t)l * D_ * D_;
        const size_t wD  = (size_t)l * D_;
        // ---------------- rel_layer ----------------
        gemm(x, np_w + wDD, np_b + wD, h, M, D_, D_, 0);
        gemm(h, q_w + wDD, q_b + wD, qb, M, D_, D_, 0);
        gemm(h, k_w + wDD, k_b + wD, kb, M, D_, D_, 0);
        gemm(h, v_w + wDD, v_b + wD, vb, M, D_, D_, 0);
        flash_attn_kernel<<<dim3(N_ / 64, H_, B_), 128, 0, stream>>>(qb, kb, vb, nmask, ao);
        gemm(ao, o_w + wDD, o_b + wD, t0, M, D_, D_, 0);
        ln(h, t0, x, n1_g + wD, n1_b + wD, M, 0);                         // x1 -> h
        gemm(h, f1_w + (size_t)l * D_ * 4 * D_, f1_b + (size_t)l * 4 * D_,
             ffh, M, D_, 4 * D_, 1);                                      // gelu
        gemm(ffh, f2_w + (size_t)l * 4 * D_ * D_, f2_b + wD, t0, M, 4 * D_, D_, 0);
        ln(x, t0, h, n2_g + wD, n2_b + wD, M, 0);                         // x2 -> x
        // ---------------- hyp_layer ----------------
        hyperedge_pool_kernel<<<dim3(B_ * E_), 64, 0, stream>>>(
            x, he_members, he_types, he_mask, et_emb + (size_t)l * 16 * 32, efi);
        gemm(efi, enc_w + (size_t)l * 544 * D_, enc_b + wD, efl, B_ * E_, 544, D_, 0);
        ln(ef, efl, nullptr, enc_g + wD, enc_bt + wD, B_ * E_, 1);        // LN(gelu(.))
        zero_kernel<<<dim3(2048), 256, 0, stream>>>(nup, BND);
        zero_kernel<<<dim3(16), 256, 0, stream>>>(cnt, (size_t)B_ * N_);
        scatter_kernel<<<dim3(B_ * E_ * A_), 128, 0, stream>>>(ef, he_members, he_mask, nup, cnt);
        build_upd_in_kernel<<<dim3(B_ * N_), 256, 0, stream>>>(x, nup, cnt, ui);
        gemm(ui, upd_w + (size_t)l * 2 * D_ * D_, upd_b + wD, ul, M, 2 * D_, D_, 0);
        ln(x, ul, nullptr, upd_g + wD, upd_bt + wD, M, 1);                // LN(gelu(.))
    }

    // ---------------- pooling head ----------------
    graph_pool_kernel<<<dim3(B_), 256, 0, stream>>>(x, nmask, gp);
    gemm(gp, pool_w, pool_b, gp2, B_, D_, D_, 1);                         // gelu
    float* out = (float*)d_out;
    ln(out + BND, gp2, nullptr, pool_g, pool_bt, B_, 0);
    copy_kernel<<<dim3(2048), 256, 0, stream>>>(out, x, BND);
}
